// ShiftedWindowAttention_14139032338912
// MI455X (gfx1250) — matrix operand order repaired and checked by reading, not compiled, not run on hardware
//
#include <hip/hip_runtime.h>
#include <hip/hip_bf16.h>

typedef __attribute__((ext_vector_type(16))) _Float16 v16h;
typedef __attribute__((ext_vector_type(8)))  _Float16 v8h;
typedef __attribute__((ext_vector_type(4)))  _Float16 v4h;
typedef __attribute__((ext_vector_type(8)))  float    v8f;
typedef unsigned int u32;
typedef __attribute__((ext_vector_type(4)))  u32      v4u;
typedef __attribute__((ext_vector_type(4)))  int      v4i;
typedef __attribute__((address_space(1)))    v4i      gv4i;   // global int4
typedef __attribute__((address_space(3)))    v4i      lv4i;   // LDS int4

// ---------------------------------------------------------------------------
// CDNA5 async global->LDS copy (ASYNCcnt-tracked, no VGPR round trip)
// Builtin signature (from clang diagnostic): (int4 AS1*, int4 AS3*, Ii, Ii)
// ---------------------------------------------------------------------------
__device__ __forceinline__ void async_ld_b128(const _Float16* g, _Float16* l) {
#if __has_builtin(__builtin_amdgcn_global_load_async_to_lds_b128)
    __builtin_amdgcn_global_load_async_to_lds_b128((gv4i*)g, (lv4i*)l, 0, 0);
#else
    unsigned laddr = (unsigned)(unsigned long long)(__attribute__((address_space(3))) void*)l;
    asm volatile("global_load_async_to_lds_b128 %0, %1, off"
                 :: "v"(laddr), "v"((unsigned long long)g) : "memory");
#endif
}

__device__ __forceinline__ void wait_async_le4() {
#if __has_builtin(__builtin_amdgcn_s_wait_asynccnt)
    __builtin_amdgcn_s_wait_asynccnt(4);
#else
    asm volatile("s_wait_asynccnt 0x4" ::: "memory");
#endif
}
__device__ __forceinline__ void wait_async_0() {
#if __has_builtin(__builtin_amdgcn_s_wait_asynccnt)
    __builtin_amdgcn_s_wait_asynccnt(0);
#else
    asm volatile("s_wait_asynccnt 0x0" ::: "memory");
#endif
}

// ---------------------------------------------------------------------------
// WMMA fragment gathers (layouts per CDNA5 ISA 7.12.2, f16 16x16x32)
// A (16x32 MxK): lane L (M=L%16): halfs h: K = 16*(h/8) + 8*(L/16) + (h%8)
//   -> with LDS row-major [M][K], two 16B loads at kbase+8*(L/16) and +16.
// B (32x16 KxN): lane L (N=L%16): halfs h: K = 16*(h/8) + 8*(L/16) + (h%8)
//   -> with LDS [N][K], two 16B loads at kbase+8*(L/16) and +16.
// ---------------------------------------------------------------------------
__device__ __forceinline__ v16h frag_a(const _Float16* rowptr, int kbase, int lane) {
    union { v4u u[2]; v16h h; } f;
    const int lh = (lane >> 4) & 1;
    f.u[0] = *(const v4u*)(rowptr + kbase + 8 * lh);
    f.u[1] = *(const v4u*)(rowptr + kbase + 16 + 8 * lh);
    return f.h;
}
__device__ __forceinline__ v16h frag_b(const _Float16* rowptr, int kbase, int lane) {
    union { v4u u[2]; v16h h; } f;
    const int lh = (lane >> 4) & 1;
    f.u[0] = *(const v4u*)(rowptr + kbase + 8 * lh);
    f.u[1] = *(const v4u*)(rowptr + kbase + 16 + 8 * lh);
    return f.h;
}

// ---------------------------------------------------------------------------
// Conversions (x4 vectorized)
// ---------------------------------------------------------------------------
// x [16,56,56,384] f32 -> xh [1024 windows][49 tokens][384] f16, fused roll(-3,-3)
__global__ __launch_bounds__(256) void cvt_x_kernel(const float* __restrict__ x,
                                                    _Float16* __restrict__ xh, int total4) {
    int idx = blockIdx.x * 256 + threadIdx.x;
    if (idx >= total4) return;
    int i = idx * 4;
    int m = i / 384, c = i - m * 384;
    int win = m / 49, n = m - win * 49;
    int b = win >> 6, wloc = win & 63;
    int wy = wloc >> 3, wx = wloc & 7;
    int ty = n / 7, tx = n - ty * 7;
    int h = (wy * 7 + ty + 3) % 56;   // shifted[hh] = x[(hh+3)%56]
    int w = (wx * 7 + tx + 3) % 56;
    float4 v = *(const float4*)(x + (((size_t)b * 56 + h) * 56 + w) * 384 + c);
    v4h o = {(_Float16)v.x, (_Float16)v.y, (_Float16)v.z, (_Float16)v.w};
    *(v4h*)(xh + i) = o;
}

__global__ __launch_bounds__(256) void cvt_w_kernel(const float* __restrict__ src,
                                                    _Float16* __restrict__ dst, int total4) {
    int idx = blockIdx.x * 256 + threadIdx.x;
    if (idx >= total4) return;
    float4 v = *(const float4*)(src + idx * 4);
    v4h o = {(_Float16)v.x, (_Float16)v.y, (_Float16)v.z, (_Float16)v.w};
    *(v4h*)(dst + idx * 4) = o;
}

// ---------------------------------------------------------------------------
// Precompute rel-pos bias + shift mask:  btab[wtype][head][qn(64)][key(64)]
// wtype = (wy==7)*2 + (wx==7)  (only edge windows see nonzero mask).
// Padded entries get -3e4 -> exp underflows to exactly 0.
// ---------------------------------------------------------------------------
__global__ __launch_bounds__(256) void prep_bias_kernel(const float* __restrict__ rel,
                                                        float* __restrict__ btab) {
    const int slot = blockIdx.x;          // 0..47 = wtype*12 + head
    const int wtype = slot / 12, head = slot - wtype * 12;
    const int ey = wtype >> 1, ex = wtype & 1;
    for (int t = threadIdx.x; t < 4096; t += 256) {
        int qn = t >> 6, key = t & 63;
        float v;
        if (qn < 49 && key < 49) {
            int ty = qn / 7, tx = qn - ty * 7;
            int my = key / 7, mx = key - my * 7;
            v = rel[((ty - my + 6) * 13 + (tx - mx + 6)) * 12 + head];
            int rq = (ey ? (ty < 4 ? 1 : 2) : 0) * 3 + (ex ? (tx < 4 ? 1 : 2) : 0);
            int rk = (ey ? (my < 4 ? 1 : 2) : 0) * 3 + (ex ? (mx < 4 ? 1 : 2) : 0);
            if (rq != rk) v -= 100.0f;
        } else {
            v = -30000.0f;
        }
        btab[(size_t)slot * 4096 + t] = v;
    }
}

// ---------------------------------------------------------------------------
// Tiled WMMA GEMM:  out[m][n] = sum_k A[m][k] * Bw[n][k] + bias[n]
// 128x128 tile, BK=32, 256 threads (8 waves), each wave = 32x64 region.
// Double-buffered: async global->LDS copies of slice k+1 overlap the 8 WMMAs
// of slice k; s_wait_asynccnt<=4 releases the current buffer (in-order
// completion per wave), barrier makes it WGP-visible.
// PROJ=false: f16 row-major output (qkv).  PROJ=true: f32 scatter to d_out
// fused with window-unpartition + roll(+3,+3).
// ---------------------------------------------------------------------------
template <bool PROJ>
__global__ __launch_bounds__(256) void gemm_kernel(const _Float16* __restrict__ A,
                                                   const _Float16* __restrict__ Bw,
                                                   const float* __restrict__ bias,
                                                   void* __restrict__ out,
                                                   int M, int N, int K) {
    __shared__ _Float16 As[2][128 * 40];  // pitch 40 halfs = 80B (16B-aligned rows)
    __shared__ _Float16 Bs[2][128 * 40];
    const int tid  = threadIdx.x;
    const int lane = tid & 31;
    const int wave = tid >> 5;
    const int mb0  = blockIdx.x * 128;
    const int nb0  = blockIdx.y * 128;
    const int wm   = (wave >> 1) * 32;
    const int wn   = (wave & 1) * 64;

    v8f acc[2][4];
#pragma unroll
    for (int i = 0; i < 2; ++i)
#pragma unroll
        for (int j = 0; j < 4; ++j)
#pragma unroll
            for (int r = 0; r < 8; ++r) acc[i][j][r] = 0.0f;

    const int lrow = tid >> 1;
    const int lseg = (tid & 1) * 16;
    const _Float16* agBase = A  + (size_t)(mb0 + lrow) * K + lseg;
    const _Float16* bgBase = Bw + (size_t)(nb0 + lrow) * K + lseg;

    // prologue: slice 0 into buffer 0 (4 async b128 per thread per slice)
    {
        _Float16* la = &As[0][lrow * 40 + lseg];
        _Float16* lb = &Bs[0][lrow * 40 + lseg];
        async_ld_b128(agBase,     la);
        async_ld_b128(agBase + 8, la + 8);
        async_ld_b128(bgBase,     lb);
        async_ld_b128(bgBase + 8, lb + 8);
    }

    int cur = 0;
    for (int k0 = 0; k0 < K; k0 += 32) {
        const bool more = (k0 + 32) < K;
        if (more) {  // issue next slice into the other buffer (freed last iter)
            _Float16* la = &As[cur ^ 1][lrow * 40 + lseg];
            _Float16* lb = &Bs[cur ^ 1][lrow * 40 + lseg];
            const _Float16* ag = agBase + k0 + 32;
            const _Float16* bg = bgBase + k0 + 32;
            async_ld_b128(ag,     la);
            async_ld_b128(ag + 8, la + 8);
            async_ld_b128(bg,     lb);
            async_ld_b128(bg + 8, lb + 8);
            wait_async_le4();   // current slice's 4 copies have landed
        } else {
            wait_async_0();
        }
        __syncthreads();
#pragma unroll
        for (int i = 0; i < 2; ++i) {
            v16h av = frag_a(As[cur] + (wm + i * 16 + (lane & 15)) * 40, 0, lane);
#pragma unroll
            for (int j = 0; j < 4; ++j) {
                v16h bv = frag_b(Bs[cur] + (wn + j * 16 + (lane & 15)) * 40, 0, lane);
                acc[i][j] = __builtin_amdgcn_wmma_f32_16x16x32_f16(
                    false, av, false, bv, (short)0, acc[i][j], false, false);
            }
        }
        __syncthreads();
        cur ^= 1;
    }

#pragma unroll
    for (int i = 0; i < 2; ++i) {
#pragma unroll
        for (int j = 0; j < 4; ++j) {
            const int n  = nb0 + wn + j * 16 + (lane & 15);
            const float bn = bias[n];
#pragma unroll
            for (int r = 0; r < 8; ++r) {
                const int m = mb0 + wm + i * 16 + ((lane >> 4) & 1) * 8 + r;
                float v = acc[i][j][r] + bn;
                if (!PROJ) {
                    ((_Float16*)out)[(size_t)m * N + n] = (_Float16)v;
                } else {
                    int win = m / 49, ntok = m - win * 49;
                    int b = win >> 6, wloc = win & 63;
                    int wy = wloc >> 3, wx = wloc & 7;
                    int ty = ntok / 7, tx = ntok - ty * 7;
                    int h = (wy * 7 + ty + 3) % 56;   // roll back (+3,+3)
                    int w = (wx * 7 + tx + 3) % 56;
                    ((float*)out)[(((size_t)b * 56 + h) * 56 + w) * 384 + n] = v;
                }
            }
        }
    }
}

// ---------------------------------------------------------------------------
// Attention: one block per (window, head); 4 waves. N=49 padded to 64, hd=32.
// ---------------------------------------------------------------------------
__global__ __launch_bounds__(128) void attn_kernel(const _Float16* __restrict__ qkv,
                                                   const float* __restrict__ btab,
                                                   _Float16* __restrict__ Y) {
    const int win  = blockIdx.x;
    const int head = blockIdx.y;
    const int tid  = threadIdx.x;
    const int lane = tid & 31;
    const int wave = tid >> 5;

    __shared__ _Float16 qs[64 * 40];   // [token][d], q pre-scaled
    __shared__ _Float16 ks[64 * 40];   // [token][d]  == B layout for S=QK^T
    __shared__ _Float16 vts[32 * 72];  // [d][token]  == B layout for P*V
    __shared__ float    s32[64 * 65];  // scores f32 (pitch 65 -> conflict-free rows)
    __shared__ _Float16 p16[64 * 72];  // probabilities f16 (A layout for P*V)

    const _Float16* base = qkv + (size_t)win * 49 * 1152 + head * 32;

    // ---- load q,k,v (16B vector loads); q pre-scaled by 32^-0.5
    for (int t = tid; t < 256; t += 128) {
        int n = t >> 2, d0 = (t & 3) * 8;
        v8h qv = {0, 0, 0, 0, 0, 0, 0, 0};
        v8h kv = qv, vv = qv;
        if (n < 49) {
            const _Float16* p = base + n * 1152 + d0;
            qv = *(const v8h*)p;
            kv = *(const v8h*)(p + 384);
            vv = *(const v8h*)(p + 768);
        }
        qv = qv * (_Float16)0.17677669529663689f;
        *(v8h*)(qs + n * 40 + d0) = qv;
        *(v8h*)(ks + n * 40 + d0) = kv;
#pragma unroll
        for (int r = 0; r < 8; ++r) vts[(d0 + r) * 72 + n] = vv[r];
    }
    __syncthreads();

    // ---- S = Q*K^T + (bias+mask) table  (K=32 -> single WMMA per tile)
    const int mb = wave * 16;
    const int wloc = win & 63;
    const int wtype = ((wloc >> 3) == 7 ? 2 : 0) + ((wloc & 7) == 7 ? 1 : 0);
    const float* bt = btab + (size_t)(wtype * 12 + head) * 4096;
    {
        v16h qf = frag_a(qs + (mb + (lane & 15)) * 40, 0, lane);
#pragma unroll
        for (int nt = 0; nt < 4; ++nt) {
            v16h kf = frag_b(ks + (nt * 16 + (lane & 15)) * 40, 0, lane);
            v8f c;
#pragma unroll
            for (int r = 0; r < 8; ++r) c[r] = 0.0f;
            c = __builtin_amdgcn_wmma_f32_16x16x32_f16(false, qf, false, kf, (short)0, c,
                                                       false, false);
            const int key = nt * 16 + (lane & 15);
#pragma unroll
            for (int r = 0; r < 8; ++r) {
                const int qn = mb + ((lane >> 4) & 1) * 8 + r;
                s32[qn * 65 + key] = c[r] + bt[(qn << 6) + key];
            }
        }
    }
    __syncthreads();

    // ---- row softmax over 49 keys (2 threads per row, shuffle combine)
    {
        const int row = tid >> 1, half = tid & 1;
        if (row < 49) {
            const int k0 = half ? 25 : 0, k1 = half ? 49 : 25;
            float mx = -3.0e4f;
            for (int m = k0; m < k1; ++m) mx = fmaxf(mx, s32[row * 65 + m]);
            mx = fmaxf(mx, __shfl_xor(mx, 1, 32));
            float sum = 0.f;
            for (int m = k0; m < k1; ++m) {
                float e = __expf(s32[row * 65 + m] - mx);
                s32[row * 65 + m] = e;
                sum += e;
            }
            sum += __shfl_xor(sum, 1, 32);
            float inv = 1.0f / sum;
            for (int m = k0; m < k1; ++m)
                p16[row * 72 + m] = (_Float16)(s32[row * 65 + m] * inv);
            if (half)
                for (int m = 49; m < 64; ++m) p16[row * 72 + m] = (_Float16)0.f;
        } else {
            for (int m = half * 32; m < half * 32 + 32; ++m)
                p16[row * 72 + m] = (_Float16)0.f;
        }
    }
    __syncthreads();

    // ---- out = P * V  (K=64 -> 2 WMMA K-steps)
#pragma unroll
    for (int nt = 0; nt < 2; ++nt) {
        v8f c;
#pragma unroll
        for (int r = 0; r < 8; ++r) c[r] = 0.0f;
#pragma unroll
        for (int kk = 0; kk < 2; ++kk) {
            v16h a = frag_a(p16 + (mb + (lane & 15)) * 72, kk * 32, lane);
            v16h b = frag_b(vts + (nt * 16 + (lane & 15)) * 72, kk * 32, lane);
            c = __builtin_amdgcn_wmma_f32_16x16x32_f16(false, a, false, b, (short)0, c,
                                                       false, false);
        }
        const int d = nt * 16 + (lane & 15);
#pragma unroll
        for (int r = 0; r < 8; ++r) {
            const int n = mb + ((lane >> 4) & 1) * 8 + r;
            if (n < 49)
                Y[((size_t)win * 49 + n) * 384 + head * 32 + d] = (_Float16)c[r];
        }
    }
}

// ---------------------------------------------------------------------------
// Launcher
// ---------------------------------------------------------------------------
extern "C" void kernel_launch(void* const* d_in, const int* in_sizes, int n_in,
                              void* d_out, int out_size, void* d_ws, size_t ws_size,
                              hipStream_t stream) {
    const float* x      = (const float*)d_in[0];
    const float* qkv_w  = (const float*)d_in[1];
    const float* qkv_b  = (const float*)d_in[2];
    const float* proj_w = (const float*)d_in[3];
    const float* proj_b = (const float*)d_in[4];
    const float* rel    = (const float*)d_in[5];

    char* ws = (char*)d_ws;
    _Float16* xh   = (_Float16*)(ws);                 // 50176*384*2 = 38,535,168
    _Float16* wq   = (_Float16*)(ws + 38535168);      // 1152*384*2  =    884,736
    _Float16* wp   = (_Float16*)(ws + 39419904);      // 384*384*2   =    294,912
    _Float16* qkvh = (_Float16*)(ws + 39714816);      // 50176*1152*2= 115,605,504
    _Float16* y    = (_Float16*)(ws + 155320320);     // 50176*384*2 =  38,535,168
    float*    btab = (float*)(ws + 193855488);        // 48*64*64*4  =    786,432

    const int M = 50176;  // 1024 windows * 49 tokens

    cvt_x_kernel<<<(M * 384 / 4 + 255) / 256, 256, 0, stream>>>(x, xh, M * 384 / 4);
    cvt_w_kernel<<<(1152 * 384 / 4 + 255) / 256, 256, 0, stream>>>(qkv_w, wq, 1152 * 384 / 4);
    cvt_w_kernel<<<(384 * 384 / 4 + 255) / 256, 256, 0, stream>>>(proj_w, wp, 384 * 384 / 4);
    prep_bias_kernel<<<48, 256, 0, stream>>>(rel, btab);

    gemm_kernel<false><<<dim3(M / 128, 1152 / 128), 256, 0, stream>>>(
        xh, wq, qkv_b, (void*)qkvh, M, 1152, 384);

    attn_kernel<<<dim3(1024, 12), 128, 0, stream>>>(qkvh, btab, y);

    gemm_kernel<true><<<dim3(M / 128, 384 / 128), 256, 0, stream>>>(
        y, wp, proj_b, d_out, M, 384, 384);
}
